// MFN_37125697307199
// MI455X (gfx1250) — compile-verified, matrix operand order checked by reading, not run on hardware
//
#include <hip/hip_runtime.h>

// ---------------------------------------------------------------------------
// MFN on MI455X (gfx1250): one workgroup (8 wave32s) owns a 16-row batch tile
// and runs the whole T=128 recurrence out of LDS. All GEMMs use
// v_wmma_f32_16x16x32_f16 with pre-packed f16 weight fragments in workspace.
// Round 3: force weight/bias loads onto the GLOBAL path (addrspace(1)) so the
// hot loop issues global_load_b128 (VMEM-only) instead of flat_load_b128
// (which also ticks DScnt and arbitrates against our ds_load A-fragments).
// ---------------------------------------------------------------------------

#define T_STEPS 128
#define NBATCH  512
#define NW      8          // waves per block (256 threads, wave32)
#define LDK     1792       // stride of the big activation staging buffer (cols)
#define LDX     576        // stride of the LSTM x||h staging buffer (cols)

typedef _Float16 v16h __attribute__((ext_vector_type(16)));
typedef _Float16 v2h  __attribute__((ext_vector_type(2)));
typedef float    v8f  __attribute__((ext_vector_type(8)));

// AMDGPU global-address-space views (guarantee global_load_* codegen)
typedef const v16h  __attribute__((address_space(1)))* gptr_v16h;
typedef const float __attribute__((address_space(1)))* gptr_f32;

__device__ __forceinline__ float frcp(float x) { return __builtin_amdgcn_rcpf(x); }

__device__ __forceinline__ float sigm(float x) {
  return frcp(1.0f + __expf(-x));                 // v_exp_f32 + v_rcp_f32
}

__device__ __forceinline__ float fast_tanh(float x) {
#if __has_builtin(__builtin_amdgcn_tanhf)
  return __builtin_amdgcn_tanhf(x);               // V_TANH_F32
#else
  float e = __expf(2.0f * x);
  return (e - 1.0f) * frcp(e + 1.0f);
#endif
}

// Load one 16x32 f16 A fragment (CDNA5 ISA 7.12.2 layout) from LDS row-major.
//   lane L: row = L&15, half = L>>4
//   vgpr pair j<4  -> K = half*8 + 2j{,+1}
//   vgpr pair j>=4 -> K = 16 + half*8 + 2(j-4){,+1}
__device__ __forceinline__
v16h load_afrag(const _Float16* __restrict__ A, int lda, int row, int hlf, int kbase)
{
  const v2h* rowp = (const v2h*)(A + (size_t)row * lda + kbase);
  v16h a;
#pragma unroll
  for (int j = 0; j < 8; ++j) {
    const int kp = (j < 4) ? (hlf * 4 + j) : (4 + hlf * 4 + j); // pair index
    v2h tpair = rowp[kp];
    a[2 * j]     = tpair[0];
    a[2 * j + 1] = tpair[1];
  }
  return a;
}

// Cooperative GEMM: out[16 x Mout] = A[16 x K] (f16, LDS, row-major, stride lda)
//                                   @ W (f16 fragment-packed in global) + bias.
// Waves round-robin over 16-wide output column tiles, two tiles at a time so a
// single A-fragment load feeds two independent WMMA accumulator chains.
__device__ __forceinline__
void gemm16(const _Float16* __restrict__ A, int lda, int K,
            const _Float16* __restrict__ fragW,
            const float* __restrict__ bias,
            float* __restrict__ out, int ldo, int Mout)
{
  const int lane = threadIdx.x & 31;
  const int wave = threadIdx.x >> 5;
  const int hlf  = lane >> 4;
  const int col  = lane & 15;          // A-row for SRC0, D-column for VDST
  const int ktiles = K >> 5;
  const int ntiles = Mout >> 4;
  const gptr_v16h Wf = (gptr_v16h)(const v16h*)fragW;
  const gptr_f32  Bf = (gptr_f32)bias;

  int tn = wave;
  // paired tiles: tn and tn+NW share every A fragment
  for (; tn + NW < ntiles; tn += 2 * NW) {
    const float b0 = Bf[tn * 16 + col];
    const float b1 = Bf[(tn + NW) * 16 + col];
    v8f acc0, acc1;
#pragma unroll
    for (int i = 0; i < 8; ++i) { acc0[i] = b0; acc1[i] = b1; }
    const gptr_v16h w0 = Wf + (size_t)tn * ktiles * 32 + lane;
    const gptr_v16h w1 = Wf + (size_t)(tn + NW) * ktiles * 32 + lane;
    for (int kt = 0; kt < ktiles; ++kt) {
      v16h a   = load_afrag(A, lda, col, hlf, kt * 32);
      v16h bb0 = w0[(size_t)kt * 32];
      v16h bb1 = w1[(size_t)kt * 32];
      acc0 = __builtin_amdgcn_wmma_f32_16x16x32_f16(false, a, false, bb0,
                                                    (short)0, acc0, false, false);
      acc1 = __builtin_amdgcn_wmma_f32_16x16x32_f16(false, a, false, bb1,
                                                    (short)0, acc1, false, false);
    }
#pragma unroll
    for (int i = 0; i < 8; ++i) {
      out[(size_t)(i + hlf * 8) * ldo + tn * 16 + col]        = acc0[i];
      out[(size_t)(i + hlf * 8) * ldo + (tn + NW) * 16 + col] = acc1[i];
    }
  }
  // remainder (also handles Mout=16 final GEMM)
  for (; tn < ntiles; tn += NW) {
    const float b0 = Bf[tn * 16 + col];
    v8f acc;
#pragma unroll
    for (int i = 0; i < 8; ++i) acc[i] = b0;
    const gptr_v16h wfp = Wf + (size_t)tn * ktiles * 32 + lane;
    for (int kt = 0; kt < ktiles; ++kt) {
      v16h a = load_afrag(A, lda, col, hlf, kt * 32);
      v16h b = wfp[(size_t)kt * 32];
      acc = __builtin_amdgcn_wmma_f32_16x16x32_f16(false, a, false, b,
                                                   (short)0, acc, false, false);
    }
#pragma unroll
    for (int i = 0; i < 8; ++i)
      out[(size_t)(i + hlf * 8) * ldo + tn * 16 + col] = acc[i];
  }
}

struct MfnParams {
  const float *x_l, *x_a, *x_v;
  const _Float16 *fr_l, *fr_a, *fr_v;
  const _Float16 *fr_a1w1, *fr_a1w2, *fr_a2w1, *fr_a2w2;
  const _Float16 *fr_g1w1, *fr_g1w2, *fr_g2w1, *fr_g2w2;
  const _Float16 *fr_ow1, *fr_ow2;
  const float *bi_l, *bi_a, *bi_v;
  const float *b_a1b1, *b_a1b2, *b_a2b1, *b_a2b2;
  const float *b_g1b1, *b_g1b2, *b_g2b1, *b_g2b2;
  const float *b_ob1, *b_ob2;
  float *out;
};

// LDS carve-out (dynamic shared): ~275 KB per workgroup, fits 320 KB/WGP.
__global__ __launch_bounds__(256) void mfn_kernel(MfnParams p)
{
  extern __shared__ __align__(16) char smem[];
  _Float16* sA   = (_Float16*)smem;            // f16 [16][1792]
  _Float16* sX   = sA + 16 * LDK;              // f16 [16][576]
  _Float16* sHid = sX + 16 * LDX;              // f16 [16][512]
  float*    sS    = (float*)(sHid + 16 * 512); // f32 [16][1536]
  float*    s_c   = sS + 16 * 1536;            // f32 [3][16*256]
  float*    s_mem = s_c + 3 * 16 * 256;        // f32 [16*256]
  float*    s_red = s_mem + 16 * 256;          // f32 [256]
  float*    s_stat= s_red + 256;               // f32 [32]
  _Float16* s_h   = (_Float16*)(s_stat + 32);  // f16 [3][16*256]

  const int tid = threadIdx.x;
  const int rowBase = blockIdx.x * 16;

  for (int i = tid; i < 16 * 256; i += 256) {
    s_c[i] = 0.f; s_c[16*256 + i] = 0.f; s_c[2*16*256 + i] = 0.f;
    s_h[i] = (_Float16)0.f; s_h[16*256 + i] = (_Float16)0.f; s_h[2*16*256 + i] = (_Float16)0.f;
    s_mem[i] = 0.f;
  }
  __syncthreads();

  const float* xs[3] = { p.x_l, p.x_a, p.x_v };
  const int Dv[3] = { 300, 74, 35 };
  const int Dp[3] = { 320, 96, 64 };           // zero-padded to multiples of 32
  const _Float16* frg[3] = { p.fr_l, p.fr_a, p.fr_v };
  const float* big[3]    = { p.bi_l, p.bi_a, p.bi_v };

  constexpr int CHAT = 16 * 512;
  constexpr int G1O  = CHAT + 16 * 256;
  constexpr int G2O  = G1O + 16 * 256;

  for (int t = 0; t < T_STEPS; ++t) {
    // ---------------- three LSTM cells ----------------
    for (int mod = 0; mod < 3; ++mod) {
      const int D = Dv[mod], DP = Dp[mod];
      const gptr_f32 xp = (gptr_f32)(xs[mod] + ((size_t)t * NBATCH + rowBase) * D);
      for (int i = tid; i < 16 * DP; i += 256) {
        int m = i / DP, k = i - m * DP;
        float v = (k < D) ? xp[(size_t)m * D + k] : 0.f;
        sX[m * LDX + k] = (_Float16)v;
      }
      for (int i = tid; i < 16 * 256; i += 256) {
        int m = i >> 8, d = i & 255;
        sX[m * LDX + DP + d] = s_h[mod * 16 * 256 + i];
      }
      __syncthreads();
      gemm16(sX, LDX, DP + 256, frg[mod], big[mod], sS, 1024, 1024);
      __syncthreads();
      for (int i = tid; i < 16 * 256; i += 256) {
        int m = i >> 8, d = i & 255;
        float ig = sS[m * 1024 + d];
        float fg = sS[m * 1024 + 256 + d];
        float gg = sS[m * 1024 + 512 + d];
        float og = sS[m * 1024 + 768 + d];
        float co = s_c[mod * 16 * 256 + i];
        sA[m * LDK + mod * 256 + d] = (_Float16)co;                 // cStar: old c
        float cn = sigm(fg) * co + sigm(ig) * fast_tanh(gg);
        s_c[mod * 16 * 256 + i] = cn;
        sA[m * LDK + 768 + mod * 256 + d] = (_Float16)cn;           // cStar: new c
        s_h[mod * 16 * 256 + i] = (_Float16)(sigm(og) * fast_tanh(cn));
      }
      __syncthreads();
    }

    // ---------------- attention MLP1 + softmax + attended ----------------
    gemm16(sA, LDK, 1536, p.fr_a1w1, p.b_a1b1, sS, 512, 512);
    __syncthreads();
    for (int i = tid; i < 16 * 512; i += 256) sHid[i] = (_Float16)fmaxf(sS[i], 0.f);
    __syncthreads();
    gemm16(sHid, 512, 512, p.fr_a1w2, p.b_a1b2, sS, 1536, 1536);   // logits
    __syncthreads();
    {
      const int m = tid >> 4, lt = tid & 15;   // 16 threads per row
      float vmax = -3.0e38f;
      for (int k = lt; k < 1536; k += 16) vmax = fmaxf(vmax, sS[m * 1536 + k]);
      s_red[m * 16 + lt] = vmax;
      __syncthreads();
      if (lt == 0) {
        float v = -3.0e38f;
        for (int i = 0; i < 16; ++i) v = fmaxf(v, s_red[m * 16 + i]);
        s_stat[m] = v;
      }
      __syncthreads();
      const float rmax = s_stat[m];
      float ps = 0.f;
      for (int k = lt; k < 1536; k += 16) ps += __expf(sS[m * 1536 + k] - rmax);
      s_red[m * 16 + lt] = ps;
      __syncthreads();
      if (lt == 0) {
        float v = 0.f;
        for (int i = 0; i < 16; ++i) v += s_red[m * 16 + i];
        s_stat[16 + m] = v;
      }
      __syncthreads();
      const float inv = frcp(s_stat[16 + m]);
      for (int k = lt; k < 1536; k += 16) {
        float att = __expf(sS[m * 1536 + k] - rmax) * inv;
        sA[m * LDK + k] = (_Float16)(att * (float)sA[m * LDK + k]); // attended
      }
    }
    for (int i = tid; i < 16 * 256; i += 256) {                    // both = [att, mem]
      int m = i >> 8, d = i & 255;
      sA[m * LDK + 1536 + d] = (_Float16)s_mem[i];
    }
    __syncthreads();

    // ---------------- attention MLP2 -> cHat ----------------
    gemm16(sA, LDK, 1536, p.fr_a2w1, p.b_a2b1, sS, 512, 512);
    __syncthreads();
    for (int i = tid; i < 16 * 512; i += 256) sHid[i] = (_Float16)fmaxf(sS[i], 0.f);
    __syncthreads();
    gemm16(sHid, 512, 512, p.fr_a2w2, p.b_a2b2, sS + CHAT, 256, 256);
    __syncthreads();
    for (int i = tid; i < 16 * 256; i += 256) sS[CHAT + i] = fast_tanh(sS[CHAT + i]);
    __syncthreads();

    // ---------------- gamma1 ----------------
    gemm16(sA, LDK, 1792, p.fr_g1w1, p.b_g1b1, sS, 512, 512);
    __syncthreads();
    for (int i = tid; i < 16 * 512; i += 256) sHid[i] = (_Float16)fmaxf(sS[i], 0.f);
    __syncthreads();
    gemm16(sHid, 512, 512, p.fr_g1w2, p.b_g1b2, sS + G1O, 256, 256);
    __syncthreads();

    // ---------------- gamma2 ----------------
    gemm16(sA, LDK, 1792, p.fr_g2w1, p.b_g2b1, sS, 512, 512);
    __syncthreads();
    for (int i = tid; i < 16 * 512; i += 256) sHid[i] = (_Float16)fmaxf(sS[i], 0.f);
    __syncthreads();
    gemm16(sHid, 512, 512, p.fr_g2w2, p.b_g2b2, sS + G2O, 256, 256);
    __syncthreads();

    // ---------------- memory update ----------------
    for (int i = tid; i < 16 * 256; i += 256) {
      float g1v = sigm(sS[G1O + i]);
      float g2v = sigm(sS[G2O + i]);
      s_mem[i] = g1v * s_mem[i] + g2v * sS[CHAT + i];
    }
    __syncthreads();
  }

  // ---------------- output MLP: [h_l h_a h_v mem](1024) -> 512 -> 1 ----------
  for (int i = tid; i < 16 * 256; i += 256) {
    int m = i >> 8, d = i & 255;
    sA[m * LDK + d]       = s_h[i];
    sA[m * LDK + 256 + d] = s_h[16 * 256 + i];
    sA[m * LDK + 512 + d] = s_h[2 * 16 * 256 + i];
    sA[m * LDK + 768 + d] = (_Float16)s_mem[i];
  }
  __syncthreads();
  gemm16(sA, LDK, 1024, p.fr_ow1, p.b_ob1, sS, 512, 512);
  __syncthreads();
  for (int i = tid; i < 16 * 512; i += 256) sHid[i] = (_Float16)fmaxf(sS[i], 0.f);
  __syncthreads();
  gemm16(sHid, 512, 512, p.fr_ow2, p.b_ob2, sS + CHAT, 16, 16);   // Mout padded to 16
  __syncthreads();
  if (tid < 16) p.out[rowBase + tid] = sS[CHAT + tid * 16];       // col 0 is the real one
}

// ---------------------------------------------------------------------------
// Weight prep: pack W (row-major [MoutValid x K*]) into f16 WMMA-B fragments.
// Optionally fuses two matrices along K (LSTM: [W_ih (padded) ; W_hh]).
// dst element order: tileN-major, then k-tile, then lane(32), then 16 halfs,
// so each lane's v16h fragment is one contiguous 32-byte read.
// ---------------------------------------------------------------------------
__global__ void prep_frag(const float* __restrict__ Wx, int KxValid, int KxPad,
                          const float* __restrict__ Wh, int Kh,
                          _Float16* __restrict__ dst, int Mout, int MoutValid)
{
  const int  ktiles = (KxPad + Kh) >> 5;
  const long total  = (long)(Mout >> 4) * ktiles * 512;
  const long idx    = (long)blockIdx.x * 256 + threadIdx.x;
  if (idx >= total) return;
  const int e    = (int)(idx & 15);
  const int lane = (int)((idx >> 4) & 31);
  const int kt   = (int)((idx >> 9) % ktiles);
  const int tn   = (int)(idx / ((long)ktiles * 512));
  const int hlf = lane >> 4, j = e >> 1, pr = e & 1;
  const int kin = (j < 4) ? (hlf * 8 + 2 * j + pr) : (16 + hlf * 8 + 2 * (j - 4) + pr);
  const int k  = kt * 32 + kin;
  const int mo = tn * 16 + (lane & 15);
  float v = 0.f;
  if (mo < MoutValid) {
    if (k < KxPad) { if (k < KxValid) v = Wx[(size_t)mo * KxValid + k]; }
    else           { v = Wh[(size_t)mo * Kh + (k - KxPad)]; }
  }
  dst[idx] = (_Float16)v;
}

__global__ void prep_bias(const float* __restrict__ a, const float* __restrict__ b,
                          float* __restrict__ dst, int n, int npad)
{
  int i = blockIdx.x * 256 + threadIdx.x;
  if (i >= npad) return;
  float v = 0.f;
  if (i < n) { v = a[i]; if (b) v += b[i]; }
  dst[i] = v;
}

extern "C" void kernel_launch(void* const* d_in, const int* in_sizes, int n_in,
                              void* d_out, int out_size, void* d_ws, size_t ws_size,
                              hipStream_t stream)
{
  (void)in_sizes; (void)n_in; (void)out_size; (void)ws_size;
  const float* x_l    = (const float*)d_in[0];
  const float* x_a    = (const float*)d_in[1];
  const float* x_v    = (const float*)d_in[2];
  const float* W_ih_l = (const float*)d_in[3];
  const float* W_hh_l = (const float*)d_in[4];
  const float* b_ih_l = (const float*)d_in[5];
  const float* b_hh_l = (const float*)d_in[6];
  const float* W_ih_a = (const float*)d_in[7];
  const float* W_hh_a = (const float*)d_in[8];
  const float* b_ih_a = (const float*)d_in[9];
  const float* b_hh_a = (const float*)d_in[10];
  const float* W_ih_v = (const float*)d_in[11];
  const float* W_hh_v = (const float*)d_in[12];
  const float* b_ih_v = (const float*)d_in[13];
  const float* b_hh_v = (const float*)d_in[14];
  const float* att1_w1 = (const float*)d_in[15];
  const float* att1_b1 = (const float*)d_in[16];
  const float* att1_w2 = (const float*)d_in[17];
  const float* att1_b2 = (const float*)d_in[18];
  const float* att2_w1 = (const float*)d_in[19];
  const float* att2_b1 = (const float*)d_in[20];
  const float* att2_w2 = (const float*)d_in[21];
  const float* att2_b2 = (const float*)d_in[22];
  const float* g1_w1 = (const float*)d_in[23];
  const float* g1_b1 = (const float*)d_in[24];
  const float* g1_w2 = (const float*)d_in[25];
  const float* g1_b2 = (const float*)d_in[26];
  const float* g2_w1 = (const float*)d_in[27];
  const float* g2_b1 = (const float*)d_in[28];
  const float* g2_w2 = (const float*)d_in[29];
  const float* g2_b2 = (const float*)d_in[30];
  const float* out_w1 = (const float*)d_in[31];
  const float* out_b1 = (const float*)d_in[32];
  const float* out_w2 = (const float*)d_in[33];
  const float* out_b2 = (const float*)d_in[34];

  // ---- workspace carve-out (f16 fragment region, then f32 bias region) ----
  _Float16* ws = (_Float16*)d_ws;
  size_t off = 0;
  auto H = [&](size_t n) { _Float16* pp = ws + off; off += n; return pp; };
  _Float16* fr_l    = H((size_t)64 * 18 * 512);  // K=576  (320 pad + 256), Mout=1024
  _Float16* fr_a    = H((size_t)64 * 11 * 512);  // K=352  (96  pad + 256)
  _Float16* fr_v    = H((size_t)64 * 10 * 512);  // K=320  (64  pad + 256)
  _Float16* fr_a1w1 = H((size_t)32 * 48 * 512);  // 1536 -> 512
  _Float16* fr_a1w2 = H((size_t)96 * 16 * 512);  // 512  -> 1536
  _Float16* fr_a2w1 = H((size_t)32 * 48 * 512);  // 1536 -> 512
  _Float16* fr_a2w2 = H((size_t)16 * 16 * 512);  // 512  -> 256
  _Float16* fr_g1w1 = H((size_t)32 * 56 * 512);  // 1792 -> 512
  _Float16* fr_g1w2 = H((size_t)16 * 16 * 512);  // 512  -> 256
  _Float16* fr_g2w1 = H((size_t)32 * 56 * 512);  // 1792 -> 512
  _Float16* fr_g2w2 = H((size_t)16 * 16 * 512);  // 512  -> 256
  _Float16* fr_ow1  = H((size_t)32 * 32 * 512);  // 1024 -> 512
  _Float16* fr_ow2  = H((size_t)1  * 16 * 512);  // 512  -> 1 (padded to 16)
  float* fp = (float*)(ws + off);
  auto F = [&](size_t n) { float* pp = fp; fp += n; return pp; };
  float* bi_l   = F(1024);
  float* bi_a   = F(1024);
  float* bi_v   = F(1024);
  float* c_a1b1 = F(512);
  float* c_a1b2 = F(1536);
  float* c_a2b1 = F(512);
  float* c_a2b2 = F(256);
  float* c_g1b1 = F(512);
  float* c_g1b2 = F(256);
  float* c_g2b1 = F(512);
  float* c_g2b2 = F(256);
  float* c_ob1  = F(512);
  float* c_ob2  = F(16);

  auto frag = [&](const float* Wx, int KxV, int KxP, const float* Wh, int Kh,
                  _Float16* dst, int Mout, int MoV) {
    long total = (long)(Mout / 16) * ((KxP + Kh) / 32) * 512;
    prep_frag<<<dim3((unsigned)((total + 255) / 256)), dim3(256), 0, stream>>>(
        Wx, KxV, KxP, Wh, Kh, dst, Mout, MoV);
  };
  auto bias = [&](const float* a, const float* b, float* dst, int n, int np) {
    prep_bias<<<dim3((unsigned)((np + 255) / 256)), dim3(256), 0, stream>>>(a, b, dst, n, np);
  };

  frag(W_ih_l, 300, 320, W_hh_l, 256, fr_l, 1024, 1024);
  frag(W_ih_a,  74,  96, W_hh_a, 256, fr_a, 1024, 1024);
  frag(W_ih_v,  35,  64, W_hh_v, 256, fr_v, 1024, 1024);
  frag(att1_w1, 1536, 1536, nullptr, 0, fr_a1w1, 512, 512);
  frag(att1_w2,  512,  512, nullptr, 0, fr_a1w2, 1536, 1536);
  frag(att2_w1, 1536, 1536, nullptr, 0, fr_a2w1, 512, 512);
  frag(att2_w2,  512,  512, nullptr, 0, fr_a2w2, 256, 256);
  frag(g1_w1, 1792, 1792, nullptr, 0, fr_g1w1, 512, 512);
  frag(g1_w2,  512,  512, nullptr, 0, fr_g1w2, 256, 256);
  frag(g2_w1, 1792, 1792, nullptr, 0, fr_g2w1, 512, 512);
  frag(g2_w2,  512,  512, nullptr, 0, fr_g2w2, 256, 256);
  frag(out_w1, 1024, 1024, nullptr, 0, fr_ow1, 512, 512);
  frag(out_w2,  512,  512, nullptr, 0, fr_ow2, 16, 1);

  bias(b_ih_l, b_hh_l, bi_l, 1024, 1024);
  bias(b_ih_a, b_hh_a, bi_a, 1024, 1024);
  bias(b_ih_v, b_hh_v, bi_v, 1024, 1024);
  bias(att1_b1, nullptr, c_a1b1, 512, 512);
  bias(att1_b2, nullptr, c_a1b2, 1536, 1536);
  bias(att2_b1, nullptr, c_a2b1, 512, 512);
  bias(att2_b2, nullptr, c_a2b2, 256, 256);
  bias(g1_b1, nullptr, c_g1b1, 512, 512);
  bias(g1_b2, nullptr, c_g1b2, 256, 256);
  bias(g2_b1, nullptr, c_g2b1, 512, 512);
  bias(g2_b2, nullptr, c_g2b2, 256, 256);
  bias(out_b1, nullptr, c_ob1, 512, 512);
  bias(out_b2, nullptr, c_ob2, 1, 16);

  MfnParams p;
  p.x_l = x_l; p.x_a = x_a; p.x_v = x_v;
  p.fr_l = fr_l; p.fr_a = fr_a; p.fr_v = fr_v;
  p.fr_a1w1 = fr_a1w1; p.fr_a1w2 = fr_a1w2; p.fr_a2w1 = fr_a2w1; p.fr_a2w2 = fr_a2w2;
  p.fr_g1w1 = fr_g1w1; p.fr_g1w2 = fr_g1w2; p.fr_g2w1 = fr_g2w1; p.fr_g2w2 = fr_g2w2;
  p.fr_ow1 = fr_ow1; p.fr_ow2 = fr_ow2;
  p.bi_l = bi_l; p.bi_a = bi_a; p.bi_v = bi_v;
  p.b_a1b1 = c_a1b1; p.b_a1b2 = c_a1b2; p.b_a2b1 = c_a2b1; p.b_a2b2 = c_a2b2;
  p.b_g1b1 = c_g1b1; p.b_g1b2 = c_g1b2; p.b_g2b1 = c_g2b1; p.b_g2b2 = c_g2b2;
  p.b_ob1 = c_ob1; p.b_ob2 = c_ob2;
  p.out = (float*)d_out;

  // Dynamic LDS: sA(57344B) + sX(18432B) + sHid(16384B) + sS(98304B)
  //            + s_c(49152B) + s_mem(16384B) + s_red(1024B) + s_stat(128B)
  //            + s_h(24576B) = 281728 bytes  (< 320 KB/WGP)
  const size_t smemBytes = 281728;
  mfn_kernel<<<dim3(NBATCH / 16), dim3(256), smemBytes, stream>>>(p);
}
